// EnsembleDynamicsNetwork_8881992368910
// MI455X (gfx1250) — matrix operand order repaired and checked by reading, not compiled, not run on hardware
//
#include <hip/hip_runtime.h>

#define NE 8
#define HD 512
#define OBS 32
#define ACT 16
#define DIN 48
#define K1 64
#define BSZ 32768
#define MT 32
#define LSTRIDE 520   // 520 halfs = 1040 B per row: 16B-aligned, bank-staggered

typedef __attribute__((ext_vector_type(16))) _Float16 v16h;
typedef __attribute__((ext_vector_type(8)))  float    v8f;

union frag_u { v16h v; uint4 u[2]; };

// ISA 16-bit A layout (16x32): lane<16 -> K 0..7 then K 16..23; lane>=16 -> K 8..15 then 24..31
__device__ __forceinline__ v16h load_afrag(const _Float16* hb, int m0, int k) {
  const int lane = threadIdx.x & 31;
  const int r = lane & 15, hi = lane >> 4;
  const _Float16* p = hb + (m0 + r) * LSTRIDE + k + hi * 8;
  frag_u t;
  t.u[0] = *(const uint4*)(p);
  t.u[1] = *(const uint4*)(p + 16);
  return t.v;
}

// ISA 16-bit B layout (32x16): lane<16 -> col=lane, K 0..7 then K 16..23; lane>=16 -> col=lane-16, K 8..15 then 24..31
// WT is transposed weights [N][KDIM] row-major -> two 16-byte runs per lane, 32 bytes apart.
template<int KDIM>
__device__ __forceinline__ v16h load_bfrag(const _Float16* __restrict__ WT, int n0, int k) {
  const int lane = threadIdx.x & 31;
  const int c = lane & 15, hi = lane >> 4;
  const _Float16* p = WT + (n0 + c) * KDIM + k + hi * 8;
  frag_u t;
  t.u[0] = *(const uint4*)(p);
  t.u[1] = *(const uint4*)(p + 16);
  return t.v;
}

// One hidden layer: h[32, KDIM] (LDS, in-place) x WT[512, KDIM] -> relu(.+bias) -> h[32, 512]
// Wave w owns cols [w*64, w*64+64) for BOTH 16-row m-tiles: B fragments loaded once per WG.
template<int KDIM>
__device__ __forceinline__ void hidden_layer(const _Float16* __restrict__ WT,
                                             const float* __restrict__ bias,
                                             _Float16* hbuf, int w, int r16, int hi) {
  const int nW = w * 64;
  float biasv[4];
  #pragma unroll
  for (int t = 0; t < 4; ++t) biasv[t] = bias[nW + t * 16 + r16];

  const v8f vz = {0.f, 0.f, 0.f, 0.f, 0.f, 0.f, 0.f, 0.f};
  v8f acc[2][4];
  #pragma unroll
  for (int m = 0; m < 2; ++m)
    #pragma unroll
    for (int t = 0; t < 4; ++t) acc[m][t] = vz;

  #pragma unroll 4
  for (int k = 0; k < KDIM; k += 32) {
    v16h a0 = load_afrag(hbuf, 0, k);
    v16h a1 = load_afrag(hbuf, 16, k);
    #pragma unroll
    for (int t = 0; t < 4; ++t) {
      v16h b = load_bfrag<KDIM>(WT, nW + t * 16, k);
      acc[0][t] = __builtin_amdgcn_wmma_f32_16x16x32_f16(
          false, a0, false, b, (short)0, acc[0][t], false, false);
      acc[1][t] = __builtin_amdgcn_wmma_f32_16x16x32_f16(
          false, a1, false, b, (short)0, acc[1][t], false, false);
    }
  }

  __syncthreads();  // all A-fragment reads done before in-place update
  #pragma unroll
  for (int m = 0; m < 2; ++m) {
    #pragma unroll
    for (int t = 0; t < 4; ++t) {
      const int col = nW + t * 16 + r16;
      #pragma unroll
      for (int rr = 0; rr < 8; ++rr) {
        float v = acc[m][t][rr] + biasv[t];
        v = v > 0.f ? v : 0.f;  // relu
        hbuf[(m * 16 + hi * 8 + rr) * LSTRIDE + col] = (_Float16)v;
      }
    }
  }
  __syncthreads();
}

__global__ void prep_x_kernel(const float* __restrict__ st, const float* __restrict__ ac,
                              const float* __restrict__ sm, const float* __restrict__ ss,
                              const float* __restrict__ am, const float* __restrict__ as_,
                              _Float16* __restrict__ xbf) {
  int i = blockIdx.x * blockDim.x + threadIdx.x;
  if (i >= BSZ * K1) return;
  int b = i >> 6, c = i & 63;
  float v = 0.0f;
  if (c < OBS)       v = (st[b * OBS + c] - sm[c]) / ss[c];
  else if (c < DIN) { int j = c - OBS; v = (ac[b * ACT + j] - am[j]) / as_[j]; }
  xbf[i] = (_Float16)v;
}

// fp32 [E,K,N] -> f16 transposed [E, N, Kpad] (zero-padded K), optional row offset for packing heads
__global__ void prep_wT_kernel(const float* __restrict__ src, _Float16* __restrict__ dst,
                               int K, int N, int Kpad, int ensStride, int rowOff) {
  int i = blockIdx.x * blockDim.x + threadIdx.x;
  if (i >= NE * N * Kpad) return;
  int k = i % Kpad;
  int n = (i / Kpad) % N;
  int e = i / (Kpad * N);
  float v = (k < K) ? src[((size_t)e * K + k) * N + n] : 0.0f;
  dst[(size_t)e * ensStride + (size_t)(rowOff + n) * Kpad + k] = (_Float16)v;
}

__global__ __launch_bounds__(256) void fused_mlp_kernel(
    const _Float16* __restrict__ xbf,
    const _Float16* __restrict__ W1T, const _Float16* __restrict__ W2T,
    const _Float16* __restrict__ W3T, const _Float16* __restrict__ W4T,
    const _Float16* __restrict__ WhT,
    const float* __restrict__ b1, const float* __restrict__ b2,
    const float* __restrict__ b3, const float* __restrict__ b4,
    const float* __restrict__ bm, const float* __restrict__ bls,
    const float* __restrict__ Wr, const float* __restrict__ br,
    const float* __restrict__ Wd, const float* __restrict__ bd,
    const float* __restrict__ dmean, const float* __restrict__ dstd,
    float* __restrict__ out)
{
  __shared__ __align__(16) _Float16 hbuf[MT * LSTRIDE];
  const int e = blockIdx.y;
  const int rowBase = blockIdx.x * MT;
  const int tid = threadIdx.x;
  const int w = tid >> 5, lane = tid & 31;
  const int r16 = lane & 15, hi = lane >> 4;

  // stage input tile: 32 rows x 64 halfs, one b128 per thread (16B aligned both sides)
  {
    const int rr = tid >> 3;        // 0..31
    const int c8 = (tid & 7) * 8;   // 0..56
    *(uint4*)(hbuf + rr * LSTRIDE + c8) =
        *(const uint4*)(xbf + (size_t)(rowBase + rr) * K1 + c8);
  }
  __syncthreads();

  hidden_layer<K1>(W1T + (size_t)e * HD * K1, b1 + e * HD, hbuf, w, r16, hi);
  hidden_layer<HD>(W2T + (size_t)e * HD * HD, b2 + e * HD, hbuf, w, r16, hi);
  hidden_layer<HD>(W3T + (size_t)e * HD * HD, b3 + e * HD, hbuf, w, r16, hi);
  hidden_layer<HD>(W4T + (size_t)e * HD * HD, b4 + e * HD, hbuf, w, r16, hi);

  // ---- mean / logstd heads: one 32x64 GEMM (cols 0..31 = mean, 32..63 = logstd) ----
  {
    const _Float16* WT = WhT + (size_t)e * 64 * HD;
    const int hm0 = (w >> 2) * 16;  // 0 or 16
    const int hn0 = (w & 3) * 16;   // 0..48
    v8f acc = {0.f, 0.f, 0.f, 0.f, 0.f, 0.f, 0.f, 0.f};
    #pragma unroll 4
    for (int k = 0; k < HD; k += 32) {
      v16h a = load_afrag(hbuf, hm0, k);
      v16h b = load_bfrag<HD>(WT, hn0, k);
      acc = __builtin_amdgcn_wmma_f32_16x16x32_f16(
          false, a, false, b, (short)0, acc, false, false);
    }
    const int col = hn0 + r16;
    #pragma unroll
    for (int rr = 0; rr < 8; ++rr) {
      const int brow = rowBase + hm0 + hi * 8 + rr;
      float v = acc[rr];
      if (col < OBS) {
        const int j = col;
        v += bm[e * OBS + j];
        out[(size_t)(e * BSZ + brow) * OBS + j] = v * dstd[j] + dmean[j];
      } else {
        const int j = col - OBS;
        v += bls[e * OBS + j];
        v = fminf(fmaxf(v, -10.0f), 0.5f);
        out[8388608 + (size_t)(e * BSZ + brow) * OBS + j] = expf(v) * dstd[j];
      }
    }
  }

  // ---- reward / done heads (N=1): plain dot products from LDS ----
  if (tid < 64) {
    const int row = tid & 31;
    const bool isD = tid >= 32;
    const float* wv = (isD ? Wd : Wr) + (size_t)e * HD;
    float s = 0.0f;
    #pragma unroll 8
    for (int k = 0; k < HD; ++k)
      s += (float)hbuf[row * LSTRIDE + k] * wv[k];
    s += (isD ? bd : br)[e];
    out[(isD ? 17039360u : 16777216u) + (size_t)e * BSZ + rowBase + row] = s;
  }
}

extern "C" void kernel_launch(void* const* d_in, const int* in_sizes, int n_in,
                              void* d_out, int out_size, void* d_ws, size_t ws_size,
                              hipStream_t stream) {
  const float* states = (const float*)d_in[0];
  const float* actions = (const float*)d_in[1];
  const float* smean = (const float*)d_in[2];
  const float* sstd  = (const float*)d_in[3];
  const float* amean = (const float*)d_in[4];
  const float* astd  = (const float*)d_in[5];
  const float* dmean = (const float*)d_in[6];
  const float* dstd  = (const float*)d_in[7];
  const float* W1 = (const float*)d_in[8];   const float* b1 = (const float*)d_in[9];
  const float* W2 = (const float*)d_in[10];  const float* b2 = (const float*)d_in[11];
  const float* W3 = (const float*)d_in[12];  const float* b3 = (const float*)d_in[13];
  const float* W4 = (const float*)d_in[14];  const float* b4 = (const float*)d_in[15];
  const float* Wm = (const float*)d_in[16];  const float* bm = (const float*)d_in[17];
  const float* Wls = (const float*)d_in[18]; const float* bls = (const float*)d_in[19];
  const float* Wr = (const float*)d_in[20];  const float* br = (const float*)d_in[21];
  const float* Wd = (const float*)d_in[22];  const float* bd = (const float*)d_in[23];
  float* out = (float*)d_out;

  char* ws = (char*)d_ws;
  _Float16* xbf = (_Float16*)(ws);                   // 32768*64*2      = 4,194,304
  _Float16* W1T = (_Float16*)(ws + 4194304);         // 8*512*64*2     =   524,288
  _Float16* W2T = (_Float16*)(ws + 4718592);         // 8*512*512*2    = 4,194,304
  _Float16* W3T = (_Float16*)(ws + 8912896);
  _Float16* W4T = (_Float16*)(ws + 13107200);
  _Float16* WhT = (_Float16*)(ws + 17301504);        // 8*64*512*2     =   524,288

  prep_x_kernel<<<(BSZ * K1 + 255) / 256, 256, 0, stream>>>(
      states, actions, smean, sstd, amean, astd, xbf);

  prep_wT_kernel<<<(NE * HD * K1 + 255) / 256, 256, 0, stream>>>(
      W1, W1T, DIN, HD, K1, HD * K1, 0);
  prep_wT_kernel<<<(NE * HD * HD + 255) / 256, 256, 0, stream>>>(
      W2, W2T, HD, HD, HD, HD * HD, 0);
  prep_wT_kernel<<<(NE * HD * HD + 255) / 256, 256, 0, stream>>>(
      W3, W3T, HD, HD, HD, HD * HD, 0);
  prep_wT_kernel<<<(NE * HD * HD + 255) / 256, 256, 0, stream>>>(
      W4, W4T, HD, HD, HD, HD * HD, 0);
  prep_wT_kernel<<<(NE * OBS * HD + 255) / 256, 256, 0, stream>>>(
      Wm, WhT, HD, OBS, HD, 64 * HD, 0);
  prep_wT_kernel<<<(NE * OBS * HD + 255) / 256, 256, 0, stream>>>(
      Wls, WhT, HD, OBS, HD, 64 * HD, 32);

  fused_mlp_kernel<<<dim3(BSZ / MT, NE), 256, 0, stream>>>(
      xbf, W1T, W2T, W3T, W4T, WhT,
      b1, b2, b3, b4, bm, bls, Wr, br, Wd, bd, dmean, dstd, out);
}
